// SOM_21509196218710
// MI455X (gfx1250) — matrix operand order repaired and checked by reading, not compiled, not run on hardware
//
#include <hip/hip_runtime.h>

// ---------------------------------------------------------------------------
// SOM-VAE forward for MI455X (gfx1250, wave32, WMMA).
// Conv3d 3x3x3 as implicit GEMM on v_wmma_f32_16x16x32_f16.
//   M = Cout (MU tiles of 16 per wave), N = B*D*H*W (4 tiles of 16 per wave),
//   K = Cin*27 (chunks of 32).
// Inputs pre-padded (+1 halo, +2 spare channels) and pre-converted to f16:
// the gather loop is branch-free, uses 32-bit offsets off the uniform base
// pointer (GVS addressing), and a loop-carried (cin,d) counter instead of
// magic division (k advance of 32 == cin+=1, d+=5 mod 27). K-overrun columns
// are zeroed in packed A fragments, not guarded in B. One B-fragment build
// feeds MU*NT WMMAs. BN batch stats fused into the conv epilogue.
// ---------------------------------------------------------------------------

typedef __attribute__((ext_vector_type(16))) _Float16 v16h;
typedef __attribute__((ext_vector_type(8)))  float    v8f;

union Frag16 { v16h v; _Float16 e[16]; };

#define NT 4  // N-tiles (of 16 positions) per wave

// ---------------------------------------------------------------------------
// Weight packing: OIDHW f32 -> f16 A-fragment order.
// Layout: wpk[(mt*KB + kb)*512 + lane*16 + h]
// A 16x32 f16 (contiguous k): lane L (g=L>>4, m=L&15), half h (v=h>>1, p=h&1):
//   k_local = 16g + h ; zero for k>=K or cout>=Cout.
// ---------------------------------------------------------------------------
__global__ void pack_w_k(const float* __restrict__ w, _Float16* __restrict__ wpk,
                         int Cout, int K, int MT, int KB) {
  int total = MT * KB * 512;
  for (int idx = blockIdx.x * blockDim.x + threadIdx.x; idx < total;
       idx += gridDim.x * blockDim.x) {
    int h    = idx & 15;
    int lane = (idx >> 4) & 31;
    int t    = idx >> 9;          // mt*KB + kb
    int kb   = t % KB;
    int mt   = t / KB;
    int g = lane >> 4, m = lane & 15;
    int v = h >> 1, p = h & 1;
    int klocal = 16 * g + h;
    int k = kb * 32 + klocal;
    int cout = mt * 16 + m;
    float val = (cout < Cout && k < K) ? w[(long)cout * K + k] : 0.0f;
    wpk[idx] = (_Float16)val;
  }
}

// f32 [B,C,D,H,W] -> f16 zero-padded [B,C+2,D+2,H+2,W+2]
__global__ void pad_f16_k(const float* __restrict__ in, _Float16* __restrict__ outp,
                          int Bn, int C, int D, int H, int W) {
  int Dp = D + 2, Hp = H + 2, Wp = W + 2, Cp = C + 2;
  long total = (long)Bn * Cp * Dp * Hp * Wp;
  for (long i = (long)blockIdx.x * blockDim.x + threadIdx.x; i < total;
       i += (long)gridDim.x * blockDim.x) {
    long t = i;
    int x = (int)(t % Wp); t /= Wp;
    int y = (int)(t % Hp); t /= Hp;
    int z = (int)(t % Dp); t /= Dp;
    int c = (int)(t % Cp); t /= Cp;
    int b = (int)t;
    float v = 0.0f;
    if (c < C && (unsigned)(x - 1) < (unsigned)W && (unsigned)(y - 1) < (unsigned)H &&
        (unsigned)(z - 1) < (unsigned)D)
      v = in[(((long)(b * C + c) * D + (z - 1)) * H + (y - 1)) * W + (x - 1)];
    outp[i] = (_Float16)v;
  }
}

__global__ void zero_k(float* p, int n) {
  int i = blockIdx.x * blockDim.x + threadIdx.x;
  if (i < n) p[i] = 0.0f;
}

// ---------------------------------------------------------------------------
// Implicit-GEMM conv3d (pad 1, stride 1) + bias, optional fused BN stats.
// Block = 1 wave. grid.x = N/(16*NT), grid.y = ceil(Cout/16)/MU.
// B fragment: lane L supplies n = L&15, k_local = (L>>4)*16 + h.
// C/D f32 16x16: VGPR i, lane L -> m = i + 8*(L>>4), n = L&15.
// All offsets 32-bit (buffers < 2^31 bytes) -> SGPR-base + VGPR-offset loads.
// ---------------------------------------------------------------------------
template <int MU>
__global__ __launch_bounds__(32)
void conv3d_wmma_k(const _Float16* __restrict__ pin, const _Float16* __restrict__ wpk,
                   const float* __restrict__ bias, float* __restrict__ out,
                   float* __restrict__ sum, float* __restrict__ sumsq,
                   int Cin, int Cout, int D, int H, int W, int doStats) {
  const int lane = threadIdx.x;
  const int g = lane >> 4, col = lane & 15;
  const int Hp = H + 2, Wp = W + 2;
  const int HWp   = Hp * Wp;
  const int slice = (D + 2) * HWp;       // one padded channel (elements)
  const int HW    = H * W;
  const int DHW   = D * HW;
  const int K  = Cin * 27;
  const int KB = (K + 31) >> 5;
  const int mtile0 = blockIdx.y * MU;

  // per-lane, per-tile 32-bit base offsets (padded input / f32 output)
  int base[NT], obase[NT];
#pragma unroll
  for (int t0 = 0; t0 < NT; ++t0) {
    unsigned t = (unsigned)blockIdx.x * (16 * NT) + t0 * 16 + col; // Ntot%(16*NT)==0
    int x = (int)(t % W); t /= W;
    int y = (int)(t % H); t /= H;
    int z = (int)(t % D); t /= D;
    int b = (int)t;
    base[t0]  = (b * (Cin + 2)) * slice + z * HWp + y * Wp + x;
    obase[t0] = b * Cout * DHW + z * HW + y * W + x;
  }

  const _Float16* wpt = wpk + (size_t)mtile0 * KB * 512;
  v8f acc[MU][NT] = {};

  // loop-carried k decomposition: k0(kb) = kb*32 + g*16 = cin0*27 + d0
  int cin0 = 0, d0 = g * 16;

  for (int kb = 0; kb < KB; ++kb) {
    v16h a[MU];
#pragma unroll
    for (int u = 0; u < MU; ++u)
      a[u] = *(const v16h*)(wpt + ((size_t)u * KB + kb) * 512 + lane * 16);
    if (kb + 1 < KB)
      __builtin_prefetch(wpt + (size_t)(kb + 1) * 512 + lane * 16, 0, 0);

    // small-range decode of d0 (< 27)
    int kz = d0 / 9;
    int r  = d0 - kz * 9;
    int ky = r / 3;
    int kx = r - ky * 3;
    int rel = cin0 * slice + kz * HWp + ky * Wp + kx;

    Frag16 bf[NT];
#pragma unroll
    for (int h = 0; h < 16; ++h) {
#pragma unroll
      for (int t = 0; t < NT; ++t) bf[t].e[h] = pin[(unsigned)(base[t] + rel)];
      // branch-free advance of (kx,ky,kz) -> rel  (cin overrun lands in the
      // 2 spare padded channels; products are zero since packed A is zero)
      if (kx == 2) {
        kx = 0;
        if (ky == 2) {
          ky = 0;
          if (kz == 2) { kz = 0; rel += slice - 2 * HWp - 2 * Wp - 2; }
          else         { ++kz;   rel += HWp - 2 * Wp - 2; }
        } else         { ++ky;   rel += Wp - 2; }
      } else           { ++kx;   ++rel; }
    }
#pragma unroll
    for (int u = 0; u < MU; ++u)
#pragma unroll
      for (int t = 0; t < NT; ++t)
        acc[u][t] = __builtin_amdgcn_wmma_f32_16x16x32_f16(
            false, a[u], false, bf[t].v, (short)0, acc[u][t], false, false);

    // advance k0 by 32 == +27 (cin) + 5 (d), with carry
    cin0 += 1;
    d0 += 5;
    if (d0 >= 27) { d0 -= 27; cin0 += 1; }
  }

#pragma unroll
  for (int u = 0; u < MU; ++u) {
#pragma unroll
    for (int i = 0; i < 8; ++i) {
      const int m = i + 8 * g;
      const int cout = (mtile0 + u) * 16 + m;
      float s = 0.0f, s2 = 0.0f;
      if (cout < Cout) {
        float bv = bias[cout];
        int coff = cout * DHW;
#pragma unroll
        for (int t = 0; t < NT; ++t) {
          float val = acc[u][t][i] + bv;
          out[(unsigned)(obase[t] + coff)] = val;
          s += val;
          s2 += val * val;
        }
      }
      if (doStats) {
        for (int dd = 8; dd >= 1; dd >>= 1) {
          s  += __shfl_xor(s,  dd, 32);
          s2 += __shfl_xor(s2, dd, 32);
        }
        if (col == 0 && cout < Cout) {
          atomicAdd(&sum[cout],   s);
          atomicAdd(&sumsq[cout], s2);
        }
      }
    }
  }
}

__global__ void bn_stats_k(const float* __restrict__ sum, const float* __restrict__ sumsq,
                           float* __restrict__ mean, float* __restrict__ invstd,
                           int C, long Ne) {
  int c = blockIdx.x * blockDim.x + threadIdx.x;
  if (c < C) {
    float m = sum[c] / (float)Ne;
    float v = sumsq[c] / (float)Ne - m * m;   // biased var (jnp default)
    mean[c] = m;
    invstd[c] = rsqrtf(v + 1e-5f);
  }
}

__global__ void bn_lrelu_k(float* __restrict__ y, const float* __restrict__ mean,
                           const float* __restrict__ invstd, int C, long DHW, long total) {
  for (long i = (long)blockIdx.x * blockDim.x + threadIdx.x; i < total;
       i += (long)gridDim.x * blockDim.x) {
    int c = (int)((i / DHW) % C);
    float v = (y[i] - mean[c]) * invstd[c];
    y[i] = v >= 0.0f ? v : 0.2f * v;
  }
}

__global__ void maxpool2_k(const float* __restrict__ in, float* __restrict__ out,
                           int BC, int Do, int Ho, int Wo) {
  long total = (long)BC * Do * Ho * Wo;
  int Wi = 2 * Wo, Hi = 2 * Ho;
  for (long i = (long)blockIdx.x * blockDim.x + threadIdx.x; i < total;
       i += (long)gridDim.x * blockDim.x) {
    long t = i;
    int x = (int)(t % Wo); t /= Wo;
    int y = (int)(t % Ho); t /= Ho;
    int z = (int)(t % Do); t /= Do;
    int bc = (int)t;
    const float* b0 = in + (((long)bc * (2 * Do) + 2 * z) * Hi + 2 * y) * Wi + 2 * x;
    const float* b1 = b0 + (long)Hi * Wi;
    float m = fmaxf(fmaxf(fmaxf(b0[0], b0[1]), fmaxf(b0[Wi], b0[Wi + 1])),
                    fmaxf(fmaxf(b1[0], b1[1]), fmaxf(b1[Wi], b1[Wi + 1])));
    out[i] = m;
  }
}

// trilinear x2 upsample, align_corners=True: pos = i*(n-1)/(2n-1)
__global__ void up2_k(const float* __restrict__ in, float* __restrict__ out,
                      int BC, int Di, int Hi, int Wi) {
  int Do = 2 * Di, Ho = 2 * Hi, Wo = 2 * Wi;
  long total = (long)BC * Do * Ho * Wo;
  for (long i = (long)blockIdx.x * blockDim.x + threadIdx.x; i < total;
       i += (long)gridDim.x * blockDim.x) {
    long t = i;
    int x = (int)(t % Wo); t /= Wo;
    int y = (int)(t % Ho); t /= Ho;
    int z = (int)(t % Do); t /= Do;
    int bc = (int)t;
    float pz = (float)z * (float)(Di - 1) / (float)(Do - 1);
    float py = (float)y * (float)(Hi - 1) / (float)(Ho - 1);
    float px = (float)x * (float)(Wi - 1) / (float)(Wo - 1);
    int z0 = (int)pz, y0 = (int)py, x0 = (int)px;
    int z1 = min(z0 + 1, Di - 1), y1 = min(y0 + 1, Hi - 1), x1 = min(x0 + 1, Wi - 1);
    float wz = pz - z0, wy = py - y0, wx = px - x0;
    const float* base = in + (long)bc * Di * Hi * Wi;
    long sH = (long)Wi, sZ = (long)Hi * Wi;
    float c000 = base[z0 * sZ + y0 * sH + x0], c001 = base[z0 * sZ + y0 * sH + x1];
    float c010 = base[z0 * sZ + y1 * sH + x0], c011 = base[z0 * sZ + y1 * sH + x1];
    float c100 = base[z1 * sZ + y0 * sH + x0], c101 = base[z1 * sZ + y0 * sH + x1];
    float c110 = base[z1 * sZ + y1 * sH + x0], c111 = base[z1 * sZ + y1 * sH + x1];
    float c00 = c000 * (1 - wx) + c001 * wx;
    float c01 = c010 * (1 - wx) + c011 * wx;
    float c10 = c100 * (1 - wx) + c101 * wx;
    float c11 = c110 * (1 - wx) + c111 * wx;
    float c0 = c00 * (1 - wy) + c01 * wy;
    float c1 = c10 * (1 - wy) + c11 * wy;
    out[i] = c0 * (1 - wz) + c1 * wz;
  }
}

// one wave per (k, b): dist[b][k] = sum_j (z[b][j] - emb[k][j])^2
__global__ __launch_bounds__(32)
void som_dist_k(const float* __restrict__ ze, const float* __restrict__ emb,
                float* __restrict__ dist) {
  int kk = blockIdx.x;      // 0..63
  int b  = blockIdx.y;      // 0..7
  int lane = threadIdx.x;
  const float* z = ze + (long)b * 1024;
  const float* e = emb + (long)kk * 1024;
  float s = 0.0f;
  for (int j = lane; j < 1024; j += 32) {
    float d = z[j] - e[j];
    s += d * d;
  }
  for (int d = 16; d >= 1; d >>= 1) s += __shfl_xor(s, d, 32);
  if (lane == 0) dist[b * 64 + kk] = s;
}

__global__ void som_argmin_k(const float* __restrict__ dist, float* __restrict__ kout,
                             int* __restrict__ kint) {
  int b = threadIdx.x;
  if (b < 8) {
    float best = dist[b * 64];
    int bi = 0;
    for (int j = 1; j < 64; ++j) {
      float v = dist[b * 64 + j];
      if (v < best) { best = v; bi = j; }   // first-min tie-break == argmin
    }
    kint[b] = bi;
    kout[b] = (float)bi;
  }
}

__global__ void som_gather_k(const float* __restrict__ emb, const int* __restrict__ kint,
                             float* __restrict__ zq) {
  int b = blockIdx.x;
  int base = kint[b] * 1024;
  for (int j = threadIdx.x; j < 1024; j += blockDim.x)
    zq[b * 1024 + j] = emb[base + j];
}

__global__ void copy_k(const float* __restrict__ in, float* __restrict__ out, long n) {
  for (long i = (long)blockIdx.x * blockDim.x + threadIdx.x; i < n;
       i += (long)gridDim.x * blockDim.x)
    out[i] = in[i];
}

// ---------------------------------------------------------------------------
// Host orchestration
// ---------------------------------------------------------------------------
extern "C" void kernel_launch(void* const* d_in, const int* in_sizes, int n_in,
                              void* d_out, int out_size, void* d_ws, size_t ws_size,
                              hipStream_t stream) {
  (void)in_sizes; (void)n_in; (void)out_size; (void)ws_size;
  const float* x   = (const float*)d_in[0];   // [8,1,64,64,64]
  const float* emb = (const float*)d_in[1];   // [8,8,1024] -> [64,1024]
  const float* P[36];
  for (int i = 0; i < 36; ++i) P[i] = (const float*)d_in[2 + i];
  // params dict order: for each sorted weight name: weight, then bias
  enum { D0W = 0, D0B, D1W1, D1B1, D1W2, D1B2, D2W1, D2B1, D2W2, D2B2,
         D3W1, D3B1, D3W2, D3B2, D4W1, D4B1, D4W2, D4B2,
         E1W1, E1B1, E1W2, E1B2, E2W1, E2B1, E2W2, E2B2,
         E3W1, E3B1, E3W2, E3B2, E4W1, E4B1, E4W2, E4B2, E5W, E5B };

  float* out = (float*)d_out;
  float* outZe   = out + 2097152;
  float* outZq   = outZe + 8192;
  float* outDist = outZq + 8192;
  float* outK    = outDist + 512;

  // workspace layout (~353 MiB)
  char* ws = (char*)d_ws;
  float*    actA = (float*)(ws);                     // 8*16*64^3 f32 = 128 MiB
  float*    actB = (float*)(ws + 134217728UL);       // 128 MiB
  _Float16* padP = (_Float16*)(ws + 268435456UL);    // 80 MiB padded-f16 stage
  float*    zeB  = (float*)(ws + 352321536UL);       // 8192 f32
  float*    stat = (float*)(ws + 352354304UL);       // sum|sumsq|mean|invstd (64 each)
  int*      kint = (int*)  (ws + 352355328UL);       // 8 ints
  _Float16* wpk  = (_Float16*)(ws + 352355584UL);    // 256 KiB shared pack region

  auto conv = [&](const float* in, int wi, int bi, int Cin, int Cout, int Dd,
                  float* o, bool st) {
    // 1) pad + f16-convert input
    int Dp = Dd + 2;
    long ptot = 8L * (Cin + 2) * Dp * Dp * Dp;
    pad_f16_k<<<(unsigned)((ptot + 255) / 256), 256, 0, stream>>>(in, padP, 8, Cin,
                                                                  Dd, Dd, Dd);
    // 2) pack weights into A-fragment order
    int K = Cin * 27, KB = (K + 31) / 32, MT = (Cout + 15) / 16;
    int tot = MT * KB * 512;
    pack_w_k<<<(tot + 255) / 256, 256, 0, stream>>>(P[wi], wpk, Cout, K, MT, KB);
    if (st) zero_k<<<1, 128, 0, stream>>>(stat, 128);
    // 3) WMMA conv (Ntot always divisible by 16*NT); 2 M-tiles/wave if Cout>=32
    long Nt = 8L * Dd * Dd * Dd;
    int MU = (Cout >= 32) ? 2 : 1;
    dim3 grid((unsigned)(Nt / (16 * NT)), (unsigned)(MT / MU));
    if (MU == 2)
      conv3d_wmma_k<2><<<grid, 32, 0, stream>>>(padP, wpk, P[bi], o, stat, stat + 64,
                                                Cin, Cout, Dd, Dd, Dd, st ? 1 : 0);
    else
      conv3d_wmma_k<1><<<grid, 32, 0, stream>>>(padP, wpk, P[bi], o, stat, stat + 64,
                                                Cin, Cout, Dd, Dd, Dd, st ? 1 : 0);
    // 4) BN stats + in-place BN+LReLU
    if (st) {
      bn_stats_k<<<1, 64, 0, stream>>>(stat, stat + 64, stat + 128, stat + 192, Cout, Nt);
      long tot2 = 8L * Cout * Dd * Dd * Dd;
      bn_lrelu_k<<<(unsigned)((tot2 + 255) / 256), 256, 0, stream>>>(
          o, stat + 128, stat + 192, Cout, (long)Dd * Dd * Dd, tot2);
    }
  };
  auto pool = [&](const float* in, float* o, int BC, int Do) {
    long tot = (long)BC * Do * Do * Do;
    maxpool2_k<<<(unsigned)((tot + 255) / 256), 256, 0, stream>>>(in, o, BC, Do, Do, Do);
  };
  auto up = [&](const float* in, float* o, int BC, int Di) {
    long tot = (long)BC * 8 * Di * Di * Di;
    up2_k<<<(unsigned)((tot + 255) / 256), 256, 0, stream>>>(in, o, BC, Di, Di, Di);
  };

  // ---------------- encoder ----------------
  conv(x,    E1W1, E1B1,  1, 16, 64, actA, true);
  conv(actA, E1W2, E1B2, 16, 16, 64, actB, true);
  pool(actB, actA, 8 * 16, 32);
  conv(actA, E2W1, E2B1, 16, 32, 32, actB, true);
  conv(actB, E2W2, E2B2, 32, 32, 32, actA, true);
  pool(actA, actB, 8 * 32, 16);
  conv(actB, E3W1, E3B1, 32, 64, 16, actA, true);
  conv(actA, E3W2, E3B2, 64, 64, 16, actB, true);
  pool(actB, actA, 8 * 64, 8);
  conv(actA, E4W1, E4B1, 64, 16, 8, actB, true);
  conv(actB, E4W2, E4B2, 16, 16, 8, actA, true);
  pool(actA, actB, 8 * 16, 4);
  conv(actB, E5W, E5B, 16, 16, 4, zeB, false);   // z_e = [8,1024]

  // ---------------- SOM ----------------
  copy_k<<<32, 256, 0, stream>>>(zeB, outZe, 8192);
  som_dist_k<<<dim3(64, 8), 32, 0, stream>>>(zeB, emb, outDist);
  som_argmin_k<<<1, 32, 0, stream>>>(outDist, outK, kint);
  som_gather_k<<<8, 256, 0, stream>>>(emb, kint, outZq);

  // ---------------- decoder (input = z_e) ----------------
  conv(zeB,  D4W1, D4B1, 16, 64, 4, actA, true);
  conv(actA, D4W2, D4B2, 64, 64, 4, actB, true);
  up(actB, actA, 8 * 64, 4);
  conv(actA, D3W1, D3B1, 64, 32, 8, actB, true);
  conv(actB, D3W2, D3B2, 32, 32, 8, actA, true);
  up(actA, actB, 8 * 32, 8);
  conv(actB, D2W1, D2B1, 32, 16, 16, actA, true);
  conv(actA, D2W2, D2B2, 16, 16, 16, actB, true);
  up(actB, actA, 8 * 16, 16);
  conv(actA, D1W1, D1B1, 16, 16, 32, actB, true);
  conv(actB, D1W2, D1B2, 16, 16, 32, actA, true);
  up(actA, actB, 8 * 16, 32);
  conv(actB, D0W, D0B, 16, 1, 64, out, false);   // recon -> d_out[0..2097151]
}
